// ExpertParallelWrapper_41987600286208
// MI455X (gfx1250) — matrix operand order repaired and checked by reading, compiled, not run on hardware
//
#include <hip/hip_runtime.h>
#include <hip/hip_bf16.h>
#include <math.h>

// ---------------- problem constants ----------------
#define T_TOK 16384   // B*S
#define HDIM  1024
#define NEXP  8
#define IDIM  2048
#define ISDIM 4096
#define CH    4096    // row chunk per (gateup,down) launch pair

// ---------------- GEMM tiling ----------------
#define BM 128
#define BN 64
#define BK 32
#define LDK 40        // padded LDS row stride (ushorts); 80B rows, 16B aligned

typedef unsigned short us16_t;
typedef us16_t us8  __attribute__((ext_vector_type(8)));
typedef us16_t us4  __attribute__((ext_vector_type(4)));
typedef __bf16 v16bf __attribute__((ext_vector_type(16)));
typedef float  v8f   __attribute__((ext_vector_type(8)));

union FragU { us8 h[2]; v16bf v; };

__device__ __forceinline__ us16_t f2bf(float f) {
    unsigned u = __float_as_uint(f);
    u += 0x7FFFu + ((u >> 16) & 1u);      // round-to-nearest-even
    return (us16_t)(u >> 16);
}

__device__ __forceinline__ v8f vzero8() {
    v8f z;
#pragma unroll
    for (int i = 0; i < 8; ++i) z[i] = 0.0f;
    return z;
}

// low 32 bits of a generic pointer to LDS == LDS byte address (flat aperture rule)
__device__ __forceinline__ unsigned lds_lo32(const void* p) {
    return (unsigned)(uintptr_t)p;
}

// gfx1250 async DMA: global -> LDS, 16 bytes per lane, tracked by ASYNCcnt
__device__ __forceinline__ void async_cp16(unsigned lds_addr, const void* g) {
    asm volatile("global_load_async_to_lds_b128 %0, %1, off"
                 :: "v"(lds_addr), "v"((unsigned long long)(uintptr_t)g)
                 : "memory");
}
__device__ __forceinline__ void wait_async0() {
    asm volatile("s_wait_asynccnt 0" ::: "memory");
}

// ---------------- fp32 -> bf16 conversion of x ----------------
__global__ __launch_bounds__(256)
void k_cvt_x(const float* __restrict__ x, us16_t* __restrict__ xb, int n4) {
    int i = blockIdx.x * blockDim.x + threadIdx.x;
    if (i >= n4) return;
    float4 v = *(const float4*)(x + (size_t)i * 4);
    us4 o;
    o[0] = f2bf(v.x); o[1] = f2bf(v.y); o[2] = f2bf(v.z); o[3] = f2bf(v.w);
    *(us4*)(xb + (size_t)i * 4) = o;
}

// ------- one-time weight prep: fp32 [K,N] -> bf16 [N,K] (transpose + convert) -------
__global__ __launch_bounds__(256)
void k_transpose_cvt(const float* __restrict__ src, us16_t* __restrict__ dst,
                     int K, int N) {
    __shared__ __align__(16) us16_t lT[64 * 72];   // [n][k], stride 72 (144B, 16B-aligned)
    int kBase = blockIdx.x * 64;
    int nBase = blockIdx.y * 64;
    int tid = threadIdx.x;
#pragma unroll
    for (int i = 0; i < 4; ++i) {
        int q  = i * 256 + tid;        // 0..1023 quads
        int k  = q >> 4;               // 0..63
        int nq = (q & 15) * 4;         // 0..60
        float4 v = *(const float4*)(src + (size_t)(kBase + k) * N + nBase + nq);
        lT[(nq + 0) * 72 + k] = f2bf(v.x);
        lT[(nq + 1) * 72 + k] = f2bf(v.y);
        lT[(nq + 2) * 72 + k] = f2bf(v.z);
        lT[(nq + 3) * 72 + k] = f2bf(v.w);
    }
    __syncthreads();
#pragma unroll
    for (int i = 0; i < 2; ++i) {
        int c  = i * 256 + tid;        // 0..511 b128 chunks
        int n  = c >> 3;
        int kc = (c & 7) * 8;
        *(us8*)(dst + (size_t)(nBase + n) * K + kBase + kc) =
            *(const us8*)(&lT[n * 72 + kc]);
    }
}

// ---------------- router: softmax + top-2 + compaction + shared gate ----------------
__global__ __launch_bounds__(256)
void k_router(const float* __restrict__ x, const float* __restrict__ gw,
              const float* __restrict__ sgw,
              int* __restrict__ cnt, int* __restrict__ rowsB,
              float* __restrict__ wtB, float* __restrict__ sg) {
    int t = blockIdx.x * blockDim.x + threadIdx.x;
    if (t >= T_TOK) return;
    const float4* xr = (const float4*)(x + (size_t)t * HDIM);

    float l[NEXP];
#pragma unroll
    for (int e = 0; e < NEXP; ++e) l[e] = 0.0f;
    float gs = 0.0f;

    for (int h4 = 0; h4 < HDIM / 4; ++h4) {
        float4 xv = xr[h4];
        float xs[4] = { xv.x, xv.y, xv.z, xv.w };
#pragma unroll
        for (int j = 0; j < 4; ++j) {
            int h = h4 * 4 + j;
            const float* g = gw + (size_t)h * NEXP;
#pragma unroll
            for (int e = 0; e < NEXP; ++e) l[e] = fmaf(xs[j], g[e], l[e]);
            gs = fmaf(xs[j], sgw[h], gs);
        }
    }

    float mx = l[0];
#pragma unroll
    for (int e = 1; e < NEXP; ++e) mx = fmaxf(mx, l[e]);
    float p[NEXP];
#pragma unroll
    for (int e = 0; e < NEXP; ++e) p[e] = __expf(l[e] - mx);

    int i1 = 0;
#pragma unroll
    for (int e = 1; e < NEXP; ++e) if (p[e] > p[i1]) i1 = e;
    int i2 = (i1 == 0) ? 1 : 0;
#pragma unroll
    for (int e = 0; e < NEXP; ++e) if (e != i1 && p[e] > p[i2]) i2 = e;

    float wsum = p[i1] + p[i2];
    float wa = p[i1] / wsum;
    float wb = p[i2] / wsum;

    int pos = atomicAdd(&cnt[i1], 1);
    rowsB[i1 * T_TOK + pos] = t;  wtB[i1 * T_TOK + pos] = wa;
    pos = atomicAdd(&cnt[i2], 1);
    rowsB[i2 * T_TOK + pos] = t;  wtB[i2 * T_TOK + pos] = wb;

    sg[t] = 1.0f / (1.0f + __expf(-gs));
}

// ---------------- fused gate+up WMMA GEMM: act = silu(x@W1) * (x@W3) ----------------
// A: xb bf16 [T,HDIM] (gathered rows).  B: w1T/w3T bf16 [Ndim, HDIM] (pre-transposed).
__global__ __launch_bounds__(256)
void k_gateup(const us16_t* __restrict__ xb,
              const us16_t* __restrict__ w1T, const us16_t* __restrict__ w3T,
              us16_t* __restrict__ act,
              const int* __restrict__ rows,      // null -> identity base+m
              const int* __restrict__ cntPtr,    // null -> full chunk
              int base, int Ndim) {
    __shared__ __align__(16) us16_t lA [2][BM * LDK];
    __shared__ __align__(16) us16_t lB1[2][BN * LDK];
    __shared__ __align__(16) us16_t lB3[2][BN * LDK];
    __shared__ int lRows[BM];

    int navail = CH;
    if (cntPtr) { navail = *cntPtr - base; if (navail > CH) navail = CH; }
    int tileM = blockIdx.x * BM;
    if (tileM >= navail) return;
    int tileN = blockIdx.y * BN;

    int tid = threadIdx.x;
    if (tid < BM) {
        int m = tileM + tid;
        lRows[tid] = (m < navail) ? (rows ? rows[base + m] : (base + m)) : 0;
    }
    __syncthreads();

    // per-thread staging descriptors (computed once; inner loop only adds kb)
    const us16_t* gA[2]; unsigned dA[2][2];
#pragma unroll
    for (int c = 0; c < 2; ++c) {
        int chunk = tid * 2 + c;       // 0..511
        int r  = chunk >> 2;           // 0..127
        int kc = (chunk & 3) * 8;      // 0,8,16,24
        gA[c] = xb + (size_t)lRows[r] * HDIM + kc;
        dA[c][0] = lds_lo32(&lA[0][r * LDK + kc]);
        dA[c][1] = lds_lo32(&lA[1][r * LDK + kc]);
    }
    int nB  = tid >> 2;                // 0..63
    int kcB = (tid & 3) * 8;
    const us16_t* gB1 = w1T + (size_t)(tileN + nB) * HDIM + kcB;
    const us16_t* gB3 = w3T + (size_t)(tileN + nB) * HDIM + kcB;
    unsigned dB1[2] = { lds_lo32(&lB1[0][nB * LDK + kcB]), lds_lo32(&lB1[1][nB * LDK + kcB]) };
    unsigned dB3[2] = { lds_lo32(&lB3[0][nB * LDK + kcB]), lds_lo32(&lB3[1][nB * LDK + kcB]) };

    auto stage = [&](int kb, int b) {
        async_cp16(dA[0][b], gA[0] + kb);
        async_cp16(dA[1][b], gA[1] + kb);
        async_cp16(dB1[b], gB1 + kb);
        async_cp16(dB3[b], gB3 + kb);
    };

    int wave = tid >> 5, lane = tid & 31;
    int laneLo = lane & 15, laneHi = lane >> 4;
    int waveM = wave & 3, waveN = wave >> 2;

    v8f acc1[2][2], acc3[2][2];
#pragma unroll
    for (int i = 0; i < 2; ++i)
#pragma unroll
        for (int j = 0; j < 2; ++j) { acc1[i][j] = vzero8(); acc3[i][j] = vzero8(); }

    stage(0, 0);
    wait_async0();
    __syncthreads();

    int cur = 0;
    for (int kb = 0; kb < HDIM; kb += BK) {
        if (kb + BK < HDIM) stage(kb + BK, cur ^ 1);   // async prefetch next tile

        FragU aF[2], b1F[2], b3F[2];
#pragma unroll
        for (int mi = 0; mi < 2; ++mi) {
            int rb = (waveM * 32 + mi * 16 + laneLo) * LDK;
            aF[mi].h[0] = *(const us8*)(&lA[cur][rb + laneHi * 8]);
            aF[mi].h[1] = *(const us8*)(&lA[cur][rb + 16 + laneHi * 8]);
        }
#pragma unroll
        for (int ni = 0; ni < 2; ++ni) {
            int cb = (waveN * 32 + ni * 16 + laneLo) * LDK + laneHi * 8;
            b1F[ni].h[0] = *(const us8*)(&lB1[cur][cb]);
            b1F[ni].h[1] = *(const us8*)(&lB1[cur][cb + 16]);
            b3F[ni].h[0] = *(const us8*)(&lB3[cur][cb]);
            b3F[ni].h[1] = *(const us8*)(&lB3[cur][cb + 16]);
        }
#pragma unroll
        for (int mi = 0; mi < 2; ++mi)
#pragma unroll
            for (int ni = 0; ni < 2; ++ni) {
                acc1[mi][ni] = __builtin_amdgcn_wmma_f32_16x16x32_bf16(
                    false, aF[mi].v, false, b1F[ni].v, (short)0, acc1[mi][ni], false, false);
                acc3[mi][ni] = __builtin_amdgcn_wmma_f32_16x16x32_bf16(
                    false, aF[mi].v, false, b3F[ni].v, (short)0, acc3[mi][ni], false, false);
            }
        wait_async0();
        __syncthreads();
        cur ^= 1;
    }

#pragma unroll
    for (int mi = 0; mi < 2; ++mi)
#pragma unroll
        for (int ni = 0; ni < 2; ++ni)
#pragma unroll
            for (int r = 0; r < 8; ++r) {
                float a1 = acc1[mi][ni][r];
                float a3 = acc3[mi][ni][r];
                float hv = (a1 / (1.0f + __expf(-a1))) * a3;
                int m = tileM + waveM * 32 + mi * 16 + laneHi * 8 + r;
                int n = tileN + waveN * 32 + ni * 16 + laneLo;
                act[(size_t)m * Ndim + n] = f2bf(hv);
            }
}

// ---------------- down-proj WMMA GEMM: out[token] += scale * (act @ W2) ----------------
// A: act bf16 [CH, Kdim].  B: w2T bf16 [HDIM, Kdim] (pre-transposed).
__global__ __launch_bounds__(256)
void k_down(const us16_t* __restrict__ act,
            const us16_t* __restrict__ w2T,
            float* __restrict__ out,
            const int* __restrict__ rows,      // null -> identity base+m
            const float* __restrict__ scales,  // indexed by base+m
            const int* __restrict__ cntPtr,    // null -> full chunk
            int base, int Kdim) {
    __shared__ __align__(16) us16_t lA[2][BM * LDK];
    __shared__ __align__(16) us16_t lB[2][BN * LDK];

    int navail = CH;
    if (cntPtr) { navail = *cntPtr - base; if (navail > CH) navail = CH; }
    int tileM = blockIdx.x * BM;
    if (tileM >= navail) return;
    int tileN = blockIdx.y * BN;   // over HDIM

    int tid = threadIdx.x;

    const us16_t* gA[2]; unsigned dA[2][2];
#pragma unroll
    for (int c = 0; c < 2; ++c) {
        int chunk = tid * 2 + c;
        int r  = chunk >> 2;
        int kc = (chunk & 3) * 8;
        gA[c] = act + (size_t)(tileM + r) * Kdim + kc;
        dA[c][0] = lds_lo32(&lA[0][r * LDK + kc]);
        dA[c][1] = lds_lo32(&lA[1][r * LDK + kc]);
    }
    int nB  = tid >> 2;
    int kcB = (tid & 3) * 8;
    const us16_t* gB = w2T + (size_t)(tileN + nB) * Kdim + kcB;
    unsigned dB[2] = { lds_lo32(&lB[0][nB * LDK + kcB]), lds_lo32(&lB[1][nB * LDK + kcB]) };

    auto stage = [&](int kb, int b) {
        async_cp16(dA[0][b], gA[0] + kb);
        async_cp16(dA[1][b], gA[1] + kb);
        async_cp16(dB[b], gB + kb);
    };

    int wave = tid >> 5, lane = tid & 31;
    int laneLo = lane & 15, laneHi = lane >> 4;
    int waveM = wave & 3, waveN = wave >> 2;

    v8f acc[2][2];
#pragma unroll
    for (int i = 0; i < 2; ++i)
#pragma unroll
        for (int j = 0; j < 2; ++j) acc[i][j] = vzero8();

    stage(0, 0);
    wait_async0();
    __syncthreads();

    int cur = 0;
    for (int kb = 0; kb < Kdim; kb += BK) {
        if (kb + BK < Kdim) stage(kb + BK, cur ^ 1);

        FragU aF[2], bF[2];
#pragma unroll
        for (int mi = 0; mi < 2; ++mi) {
            int rb = (waveM * 32 + mi * 16 + laneLo) * LDK;
            aF[mi].h[0] = *(const us8*)(&lA[cur][rb + laneHi * 8]);
            aF[mi].h[1] = *(const us8*)(&lA[cur][rb + 16 + laneHi * 8]);
        }
#pragma unroll
        for (int ni = 0; ni < 2; ++ni) {
            int cb = (waveN * 32 + ni * 16 + laneLo) * LDK + laneHi * 8;
            bF[ni].h[0] = *(const us8*)(&lB[cur][cb]);
            bF[ni].h[1] = *(const us8*)(&lB[cur][cb + 16]);
        }
#pragma unroll
        for (int mi = 0; mi < 2; ++mi)
#pragma unroll
            for (int ni = 0; ni < 2; ++ni)
                acc[mi][ni] = __builtin_amdgcn_wmma_f32_16x16x32_bf16(
                    false, aF[mi].v, false, bF[ni].v, (short)0, acc[mi][ni], false, false);
        wait_async0();
        __syncthreads();
        cur ^= 1;
    }

#pragma unroll
    for (int mi = 0; mi < 2; ++mi)
#pragma unroll
        for (int ni = 0; ni < 2; ++ni)
#pragma unroll
            for (int r = 0; r < 8; ++r) {
                int m = tileM + waveM * 32 + mi * 16 + laneHi * 8 + r;
                if (m < navail) {
                    int token = rows ? rows[base + m] : (base + m);
                    float sc = scales[base + m];
                    int n = tileN + waveN * 32 + ni * 16 + laneLo;
                    size_t oi = (size_t)token * HDIM + n;
                    out[oi] = out[oi] + sc * acc[mi][ni][r];
                }
            }
}

// ---------------- launch ----------------
extern "C" void kernel_launch(void* const* d_in, const int* in_sizes, int n_in,
                              void* d_out, int out_size, void* d_ws, size_t ws_size,
                              hipStream_t stream) {
    const float* x   = (const float*)d_in[0];
    const float* gw  = (const float*)d_in[1];
    const float* w1  = (const float*)d_in[2];
    const float* w3  = (const float*)d_in[3];
    const float* w2  = (const float*)d_in[4];
    const float* sw1 = (const float*)d_in[5];
    const float* sw3 = (const float*)d_in[6];
    const float* sw2 = (const float*)d_in[7];
    const float* sgw = (const float*)d_in[8];
    float* out = (float*)d_out;

    char* ws = (char*)d_ws;
    size_t o = 0;
    us16_t* xb   = (us16_t*)(ws + o); o += (size_t)T_TOK * HDIM * 2;    // 33.5 MB
    us16_t* act  = (us16_t*)(ws + o); o += (size_t)CH * ISDIM * 2;      // 33.5 MB
    us16_t* w1T  = (us16_t*)(ws + o); o += (size_t)NEXP * IDIM * HDIM * 2;  // 33.5 MB [E][I][H]
    us16_t* w3T  = (us16_t*)(ws + o); o += (size_t)NEXP * IDIM * HDIM * 2;  // 33.5 MB
    us16_t* w2T  = (us16_t*)(ws + o); o += (size_t)NEXP * HDIM * IDIM * 2;  // 33.5 MB [E][H][I]
    us16_t* sw1T = (us16_t*)(ws + o); o += (size_t)ISDIM * HDIM * 2;    // 8.4 MB [IS][H]
    us16_t* sw3T = (us16_t*)(ws + o); o += (size_t)ISDIM * HDIM * 2;    // 8.4 MB
    us16_t* sw2T = (us16_t*)(ws + o); o += (size_t)HDIM * ISDIM * 2;    // 8.4 MB [H][IS]
    int*    rowsB= (int*)(ws + o);    o += (size_t)NEXP * T_TOK * 4;
    float*  wtB  = (float*)(ws + o);  o += (size_t)NEXP * T_TOK * 4;
    float*  sg   = (float*)(ws + o);  o += (size_t)T_TOK * 4;
    int*    cnt  = (int*)(ws + o);    o += 256;

    hipMemsetAsync(out, 0, (size_t)T_TOK * HDIM * 4, stream);
    hipMemsetAsync(cnt, 0, NEXP * 4, stream);

    int n4 = T_TOK * HDIM / 4;
    k_cvt_x<<<(n4 + 255) / 256, 256, 0, stream>>>(x, xb, n4);
    k_router<<<T_TOK / 256, 256, 0, stream>>>(x, gw, sgw, cnt, rowsB, wtB, sg);

    // one-time weight transpose+convert (bf16, [N,K])
    for (int e = 0; e < NEXP; ++e) {
        size_t wo = (size_t)e * HDIM * IDIM;
        k_transpose_cvt<<<dim3(HDIM / 64, IDIM / 64), 256, 0, stream>>>(w1 + wo, w1T + wo, HDIM, IDIM);
        k_transpose_cvt<<<dim3(HDIM / 64, IDIM / 64), 256, 0, stream>>>(w3 + wo, w3T + wo, HDIM, IDIM);
        k_transpose_cvt<<<dim3(IDIM / 64, HDIM / 64), 256, 0, stream>>>(w2 + wo, w2T + wo, IDIM, HDIM);
    }
    k_transpose_cvt<<<dim3(HDIM / 64, ISDIM / 64), 256, 0, stream>>>(sw1, sw1T, HDIM, ISDIM);
    k_transpose_cvt<<<dim3(HDIM / 64, ISDIM / 64), 256, 0, stream>>>(sw3, sw3T, HDIM, ISDIM);
    k_transpose_cvt<<<dim3(ISDIM / 64, HDIM / 64), 256, 0, stream>>>(sw2, sw2T, ISDIM, HDIM);

    // routed experts: sparse (compacted) rows, worst-case grid + early exit
    for (int e = 0; e < NEXP; ++e) {
        const us16_t* we1 = w1T + (size_t)e * IDIM * HDIM;
        const us16_t* we3 = w3T + (size_t)e * IDIM * HDIM;
        const us16_t* we2 = w2T + (size_t)e * HDIM * IDIM;
        const int*    rB  = rowsB + (size_t)e * T_TOK;
        const float*  wB  = wtB   + (size_t)e * T_TOK;
        for (int c = 0; c < T_TOK / CH; ++c) {
            int base = c * CH;
            dim3 g1(CH / BM, IDIM / BN);
            k_gateup<<<g1, 256, 0, stream>>>(xb, we1, we3, act, rB, cnt + e, base, IDIM);
            dim3 g2(CH / BM, HDIM / BN);
            k_down<<<g2, 256, 0, stream>>>(act, we2, out, rB, wB, cnt + e, base, IDIM);
        }
    }

    // shared expert: dense identity rows, sigmoid gate scale
    for (int c = 0; c < T_TOK / CH; ++c) {
        int base = c * CH;
        dim3 g1(CH / BM, ISDIM / BN);
        k_gateup<<<g1, 256, 0, stream>>>(xb, sw1T, sw3T, act, nullptr, nullptr, base, ISDIM);
        dim3 g2(CH / BM, HDIM / BN);
        k_down<<<g2, 256, 0, stream>>>(act, sw2T, out, nullptr, sg, nullptr, base, ISDIM);
    }
}
